// MTPModule_71751723647545
// MI455X (gfx1250) — matrix operand order repaired and checked by reading, not compiled, not run on hardware
//
#include <hip/hip_runtime.h>
#include <hip/hip_bf16.h>

typedef __bf16 bf16_t;
typedef __attribute__((ext_vector_type(16))) __bf16 v16bf;
typedef __attribute__((ext_vector_type(8)))  __bf16 v8bf;
typedef __attribute__((ext_vector_type(8)))  float  v8f;

// ---------------------------------------------------------------------------
// fp32 -> bf16 elementwise convert (weights, done fresh every call).
// Source is read exactly once per call -> non-temporal loads keep L2 for the
// bf16 copies, which ARE reused across all 4 heads.
// ---------------------------------------------------------------------------
__global__ __launch_bounds__(256) void f32_to_bf16_kernel(
    const float* __restrict__ in, bf16_t* __restrict__ out, long long n) {
  long long i = (long long)blockIdx.x * 256 + threadIdx.x;
  long long stride = (long long)gridDim.x * 256;
  for (; i < n; i += stride) {
    float v = __builtin_nontemporal_load(in + i);
    out[i] = (bf16_t)v;
  }
}

// ---------------------------------------------------------------------------
// WMMA fragment loaders (layouts per CDNA5 ISA §7.12.2, wave32)
// A: 16x32 bf16, row-major source A[M,K] (ld = K)
//   lanes 0-15 : M=lane,    K = k0+0..7  (elems 0..7),  k0+16..23 (elems 8..15)
//   lanes16-31 : M=lane-16, K = k0+8..15 (elems 0..7),  k0+24..31 (elems 8..15)
// ---------------------------------------------------------------------------
__device__ __forceinline__ v16bf load_a_frag(const bf16_t* __restrict__ A,
                                             int lda, int tileM, int k0, int lane) {
  const int mrow = lane & 15;
  const int koff = (lane >> 4) ? 8 : 0;
  const bf16_t* p = A + (size_t)(tileM + mrow) * lda + k0 + koff;
  v8bf lo = *reinterpret_cast<const v8bf*>(p);
  v8bf hi = *reinterpret_cast<const v8bf*>(p + 16);
  v16bf r;
#pragma unroll
  for (int i = 0; i < 8; ++i) { r[i] = lo[i]; r[i + 8] = hi[i]; }
  return r;
}

// B: 32x16 bf16 from W[N,K] row-major (computing A @ W^T, NT layout):
//   lanes 0-15 : N=lane,    K = k0+0..7  (elems 0..7),  k0+16..23 (elems 8..15)
//   lanes16-31 : N=lane-16, K = k0+8..15 (elems 0..7),  k0+24..31 (elems 8..15)
__device__ __forceinline__ v16bf load_b_frag(const bf16_t* __restrict__ W,
                                             int ldw, int tileN, int k0, int lane) {
  const int ncol = lane & 15;
  const int koff = (lane >> 4) ? 8 : 0;
  const bf16_t* p = W + (size_t)(tileN + ncol) * ldw + k0 + koff;
  v8bf lo = *reinterpret_cast<const v8bf*>(p);
  v8bf hi = *reinterpret_cast<const v8bf*>(p + 16);
  v16bf r;
#pragma unroll
  for (int i = 0; i < 8; ++i) { r[i] = lo[i]; r[i + 8] = hi[i]; }
  return r;
}

// ---------------------------------------------------------------------------
// Generic bf16 WMMA GEMM:  C[M,N] = A[M,K] @ W[N,K]^T + bias, optional ReLU.
// Wave computes a 32x64 C tile (2 A-frags x 4 B-frags -> 8 WMMAs / K-step);
// 8 waves/block cover 256 rows. Epilogue writes fp32 (strideF, optionally
// non-temporal) and/or bf16 (strideB).
// Requires M%256==0, N%64==0, K%32==0 (true for all call sites here).
// ---------------------------------------------------------------------------
__global__ __launch_bounds__(256) void gemm_bf16_wmma(
    const bf16_t* __restrict__ A, const bf16_t* __restrict__ W,
    const float* __restrict__ bias,
    float* __restrict__ outF, long long strideF,
    bf16_t* __restrict__ outB, long long strideB,
    int M, int N, int K, int do_relu, int nt_store) {
  const int lane  = threadIdx.x & 31;
  const int wave  = threadIdx.x >> 5;
  const int tileM = (blockIdx.y * 8 + wave) * 32;
  const int tileN = blockIdx.x * 64;
  if (tileM >= M) return;

  v8f acc[8] = {v8f{}, v8f{}, v8f{}, v8f{}, v8f{}, v8f{}, v8f{}, v8f{}};

  for (int k0 = 0; k0 < K; k0 += 32) {
    v16bf a0 = load_a_frag(A, K, tileM,      k0, lane);
    v16bf a1 = load_a_frag(A, K, tileM + 16, k0, lane);
    v16bf b0 = load_b_frag(W, K, tileN +  0, k0, lane);
    v16bf b1 = load_b_frag(W, K, tileN + 16, k0, lane);
    v16bf b2 = load_b_frag(W, K, tileN + 32, k0, lane);
    v16bf b3 = load_b_frag(W, K, tileN + 48, k0, lane);
    acc[0] = __builtin_amdgcn_wmma_f32_16x16x32_bf16(false, a0, false, b0, (short)0, acc[0], false, false);
    acc[1] = __builtin_amdgcn_wmma_f32_16x16x32_bf16(false, a0, false, b1, (short)0, acc[1], false, false);
    acc[2] = __builtin_amdgcn_wmma_f32_16x16x32_bf16(false, a0, false, b2, (short)0, acc[2], false, false);
    acc[3] = __builtin_amdgcn_wmma_f32_16x16x32_bf16(false, a0, false, b3, (short)0, acc[3], false, false);
    acc[4] = __builtin_amdgcn_wmma_f32_16x16x32_bf16(false, a1, false, b0, (short)0, acc[4], false, false);
    acc[5] = __builtin_amdgcn_wmma_f32_16x16x32_bf16(false, a1, false, b1, (short)0, acc[5], false, false);
    acc[6] = __builtin_amdgcn_wmma_f32_16x16x32_bf16(false, a1, false, b2, (short)0, acc[6], false, false);
    acc[7] = __builtin_amdgcn_wmma_f32_16x16x32_bf16(false, a1, false, b3, (short)0, acc[7], false, false);
  }

  // C layout: VGPR v -> row base + v (+8 for lanes 16-31), col = tileN + j*16 + (lane&15)
  const int ccol = lane & 15;
#pragma unroll
  for (int half = 0; half < 2; ++half) {
    const int rbase = tileM + half * 16 + ((lane >> 4) ? 8 : 0);
#pragma unroll
    for (int j = 0; j < 4; ++j) {
      const int col = tileN + j * 16 + ccol;
      const float bv = bias ? bias[col] : 0.0f;
      const v8f a = acc[half * 4 + j];
#pragma unroll
      for (int v = 0; v < 8; ++v) {
        const int row = rbase + v;
        float val = a[v] + bv;
        if (do_relu) val = fmaxf(val, 0.0f);
        if (outF) {
          float* p = outF + (long long)row * strideF + col;
          if (nt_store) __builtin_nontemporal_store(val, p);
          else          *p = val;
        }
        if (outB) outB[(long long)row * strideB + col] = (bf16_t)val;
      }
    }
  }
}

// ---------------------------------------------------------------------------
// Block-wide sum reduction (256 threads)
// ---------------------------------------------------------------------------
__device__ __forceinline__ float block_sum(float v, float* sdata) {
  const int tid = threadIdx.x;
  sdata[tid] = v;
  __syncthreads();
  for (int s = 128; s > 0; s >>= 1) {
    if (tid < s) sdata[tid] += sdata[tid + s];
    __syncthreads();
  }
  float r = sdata[0];
  __syncthreads();
  return r;
}

// ---------------------------------------------------------------------------
// Stage 1: xin_bf[m, 0:1024]   = rms(emb_table[token_ids[b, 1+h+r]])
//          xin_bf[m, 1024:2048]= rms(h_prev[m])   (row m = b*256 + r)
// ---------------------------------------------------------------------------
__global__ __launch_bounds__(256) void rms_concat_kernel(
    const int* __restrict__ tok, const float* __restrict__ emb,
    const float* __restrict__ hid, const float* __restrict__ xprev, int use_hid,
    const float* __restrict__ rw, bf16_t* __restrict__ xin, int h) {
  __shared__ float sdata[256];
  const int m = blockIdx.x, b = m >> 8, r = m & 255, tid = threadIdx.x;
  const int t = tok[b * 260 + 1 + h + r];
  const float* src0 = emb + (size_t)t * 1024;
  const float* src1 = use_hid ? hid + ((size_t)b * 260 + r) * 1024
                              : xprev + (size_t)m * 1024;
#pragma unroll
  for (int part = 0; part < 2; ++part) {
    const float* src = part ? src1 : src0;
    float vals[4];
    float ss = 0.0f;
#pragma unroll
    for (int i = 0; i < 4; ++i) {
      vals[i] = src[tid + i * 256];
      ss += vals[i] * vals[i];
    }
    ss = block_sum(ss, sdata);
    const float scale = rsqrtf(ss * (1.0f / 1024.0f) + 1e-6f);
    bf16_t* dst = xin + (size_t)m * 2048 + part * 1024;
#pragma unroll
    for (int i = 0; i < 4; ++i) {
      const int d = tid + i * 256;
      dst[d] = (bf16_t)(vals[i] * scale * rw[d]);
    }
  }
}

// ---------------------------------------------------------------------------
// Residual + LayerNorm:  x = ln(x + t) * w + b ; writes fp32 (in-place) + bf16
// ---------------------------------------------------------------------------
__global__ __launch_bounds__(256) void ln_residual_kernel(
    const float* __restrict__ t, const float* __restrict__ w,
    const float* __restrict__ bb, float* __restrict__ x,
    bf16_t* __restrict__ xbf) {
  __shared__ float sdata[256];
  const int m = blockIdx.x, tid = threadIdx.x;
  float s[4];
  float sum = 0.0f;
#pragma unroll
  for (int i = 0; i < 4; ++i) {
    const int d = tid + i * 256;
    s[i] = x[(size_t)m * 1024 + d] + t[(size_t)m * 1024 + d];
    sum += s[i];
  }
  const float mu = block_sum(sum, sdata) * (1.0f / 1024.0f);
  float vs = 0.0f;
#pragma unroll
  for (int i = 0; i < 4; ++i) { const float dv = s[i] - mu; vs += dv * dv; }
  const float var = block_sum(vs, sdata) * (1.0f / 1024.0f);
  const float inv = rsqrtf(var + 1e-5f);
#pragma unroll
  for (int i = 0; i < 4; ++i) {
    const int d = tid + i * 256;
    const float o = (s[i] - mu) * inv * w[d] + bb[d];
    x[(size_t)m * 1024 + d] = o;
    xbf[(size_t)m * 1024 + d] = (bf16_t)o;
  }
}

// ---------------------------------------------------------------------------
// Host launcher
// ---------------------------------------------------------------------------
extern "C" void kernel_launch(void* const* d_in, const int* in_sizes, int n_in,
                              void* d_out, int out_size, void* d_ws, size_t ws_size,
                              hipStream_t stream) {
  (void)in_sizes; (void)n_in; (void)out_size; (void)ws_size;
  constexpr int H = 4, D = 1024, DFF = 2048, V = 32000, M = 512;

  const int*   tok     = (const int*)  d_in[0];
  const float* hid     = (const float*)d_in[1];
  const float* emb     = (const float*)d_in[2];
  const float* rms_w   = (const float*)d_in[3];
  const float* proj_w  = (const float*)d_in[4];
  const float* proj_b  = (const float*)d_in[5];
  const float* sa_wv   = (const float*)d_in[6];
  const float* sa_bv   = (const float*)d_in[7];
  const float* sa_wo   = (const float*)d_in[8];
  const float* sa_bo   = (const float*)d_in[9];
  const float* ca_wv   = (const float*)d_in[10];
  const float* ca_bv   = (const float*)d_in[11];
  const float* ca_wo   = (const float*)d_in[12];
  const float* ca_bo   = (const float*)d_in[13];
  const float* ln1_w   = (const float*)d_in[14];
  const float* ln1_b   = (const float*)d_in[15];
  const float* ln2_w   = (const float*)d_in[16];
  const float* ln2_b   = (const float*)d_in[17];
  const float* ln3_w   = (const float*)d_in[18];
  const float* ln3_b   = (const float*)d_in[19];
  const float* ff_w1   = (const float*)d_in[20];
  const float* ff_b1   = (const float*)d_in[21];
  const float* ff_w2   = (const float*)d_in[22];
  const float* ff_b2   = (const float*)d_in[23];
  const float* unemb_w = (const float*)d_in[24];
  const float* unemb_b = (const float*)d_in[25];
  float* out = (float*)d_out;

  // ---- workspace carve-up (256B aligned) ----
  char* base = (char*)d_ws;
  size_t off = 0;
  auto alloc = [&](size_t bytes) -> void* {
    void* p = base + off;
    off = (off + bytes + 255) & ~(size_t)255;
    return p;
  };
  bf16_t* wbf_proj  = (bf16_t*)alloc((size_t)H * D * 2 * D * 2);
  bf16_t* wbf_sa_wv = (bf16_t*)alloc((size_t)H * D * D * 2);
  bf16_t* wbf_sa_wo = (bf16_t*)alloc((size_t)H * D * D * 2);
  bf16_t* wbf_ca_wv = (bf16_t*)alloc((size_t)H * D * D * 2);
  bf16_t* wbf_ca_wo = (bf16_t*)alloc((size_t)H * D * D * 2);
  bf16_t* wbf_ff1   = (bf16_t*)alloc((size_t)H * DFF * D * 2);
  bf16_t* wbf_ff2   = (bf16_t*)alloc((size_t)H * D * DFF * 2);
  bf16_t* wbf_unemb = (bf16_t*)alloc((size_t)V * D * 2);
  bf16_t* xin_bf    = (bf16_t*)alloc((size_t)M * 2 * D * 2);
  bf16_t* x_bf      = (bf16_t*)alloc((size_t)M * D * 2);
  bf16_t* v_bf      = (bf16_t*)alloc((size_t)M * DFF * 2);
  float*  x_f32     = (float*) alloc((size_t)M * D * 4);
  float*  t_f32     = (float*) alloc((size_t)M * D * 4);

  auto conv = [&](const float* src, bf16_t* dst, long long n) {
    f32_to_bf16_kernel<<<2048, 256, 0, stream>>>(src, dst, n);
  };
  conv(proj_w,  wbf_proj,  (long long)H * D * 2 * D);
  conv(sa_wv,   wbf_sa_wv, (long long)H * D * D);
  conv(sa_wo,   wbf_sa_wo, (long long)H * D * D);
  conv(ca_wv,   wbf_ca_wv, (long long)H * D * D);
  conv(ca_wo,   wbf_ca_wo, (long long)H * D * D);
  conv(ff_w1,   wbf_ff1,   (long long)H * DFF * D);
  conv(ff_w2,   wbf_ff2,   (long long)H * D * DFF);
  conv(unemb_w, wbf_unemb, (long long)V * D);

  auto gemm = [&](const bf16_t* A, const bf16_t* W, const float* bias,
                  float* oF, long long sF, bf16_t* oB, long long sB,
                  int m, int n, int k, int relu, int nt) {
    dim3 grid(n / 64, m / 256);
    gemm_bf16_wmma<<<grid, 256, 0, stream>>>(A, W, bias, oF, sF, oB, sB,
                                             m, n, k, relu, nt);
  };

  for (int h = 0; h < H; ++h) {
    // x_in = concat(rms(emb_window), rms(h_prev))
    rms_concat_kernel<<<M, 256, 0, stream>>>(tok, emb, hid, x_f32, h == 0,
                                             rms_w, xin_bf, h);
    // proj: (512x2048)@(2048x1024) -> x (fp32 + bf16)
    gemm(xin_bf, wbf_proj + (size_t)h * D * 2 * D, proj_b + h * D,
         x_f32, D, x_bf, D, M, D, 2 * D, 0, 0);
    // "self-attn" block: v = x@wv^T+bv ; t = v@wo^T+bo ; x = ln(x+t)
    gemm(x_bf, wbf_sa_wv + (size_t)h * D * D, sa_bv + h * D,
         nullptr, 0, v_bf, D, M, D, D, 0, 0);
    gemm(v_bf, wbf_sa_wo + (size_t)h * D * D, sa_bo + h * D,
         t_f32, D, nullptr, 0, M, D, D, 0, 0);
    ln_residual_kernel<<<M, 256, 0, stream>>>(t_f32, ln1_w + h * D, ln1_b + h * D,
                                              x_f32, x_bf);
    // "cross-attn" block
    gemm(x_bf, wbf_ca_wv + (size_t)h * D * D, ca_bv + h * D,
         nullptr, 0, v_bf, D, M, D, D, 0, 0);
    gemm(v_bf, wbf_ca_wo + (size_t)h * D * D, ca_bo + h * D,
         t_f32, D, nullptr, 0, M, D, D, 0, 0);
    ln_residual_kernel<<<M, 256, 0, stream>>>(t_f32, ln2_w + h * D, ln2_b + h * D,
                                              x_f32, x_bf);
    // FFN: u = relu(x@w1^T+b1) ; t = u@w2^T+b2 ; x = ln(x+t)
    gemm(x_bf, wbf_ff1 + (size_t)h * DFF * D, ff_b1 + h * DFF,
         nullptr, 0, v_bf, DFF, M, DFF, D, 1, 0);
    gemm(v_bf, wbf_ff2 + (size_t)h * D * DFF, ff_b2 + h * D,
         t_f32, D, nullptr, 0, M, D, DFF, 0, 0);
    ln_residual_kernel<<<M, 256, 0, stream>>>(t_f32, ln3_w + h * D, ln3_b + h * D,
                                              x_f32, x_bf);
    // unembedding straight into d_out: out[(m*4+h)*V + col], NT stores so the
    // 262MB logits stream does not evict the 65MB bf16 unemb weights from L2.
    gemm(x_bf, wbf_unemb, unemb_b,
         out + (size_t)h * V, (long long)H * V, nullptr, 0, M, V, D, 0, 1);
  }
}
